// SparseExpertBank_57913339019894
// MI455X (gfx1250) — matrix operand order repaired and checked by reading, not compiled, not run on hardware
//
#include <hip/hip_runtime.h>
#include <hip/hip_bf16.h>

// ---------------------------------------------------------------------------
// Per-expert SwiGLU FFN, fused: (E,C,D) x {Wg,Wu:(E,DF,D), Wd:(E,D,DF)} -> (E,C,D)
// HBM-bound (1.6 GB fp32 weights, 51.5 GFLOP). Stream fp32 weights at full
// bandwidth, pack-convert to bf16 (v_add + v_perm_b32, 1.5 VALU/elem), and run
// v_wmma_f32_16x16x32_bf16 with f32 accumulation. One WG per (expert, 32-row
// half), 8 waves, 256 threads. Phase-A K-loop unrolled x4 for load-clause
// depth; hidden chunk double-buffered in LDS (1 barrier per chunk).
// ---------------------------------------------------------------------------

typedef __attribute__((ext_vector_type(16))) __bf16 v16bf;
typedef __attribute__((ext_vector_type(8)))  __bf16 v8bf;
typedef __attribute__((ext_vector_type(8)))  float  v8f;
typedef __attribute__((ext_vector_type(8)))  int    v8i;
typedef __attribute__((ext_vector_type(2)))  int    v2i;

static constexpr int kE  = 64;
static constexpr int kC  = 64;
static constexpr int kD  = 1024;
static constexpr int kDF = 2048;
static constexpr int kMT = 32;    // rows per workgroup (C split in 2)
static constexpr int kFC = 128;   // DF chunk per iteration (8 waves * 16 cols)
static constexpr int kXS = 1032;  // padded LDS row stride (elements) for X tile
static constexpr int kHS = 136;   // padded LDS row stride for hidden chunk

// Pack two f32 -> two bf16 in one u32 (a -> low 16, b -> high 16).
__device__ __forceinline__ unsigned pk_bf16(float a, float b) {
  unsigned ua = __builtin_bit_cast(unsigned, a) + 0x8000u;  // cheap nearest
  unsigned ub = __builtin_bit_cast(unsigned, b) + 0x8000u;
  // bytes: dst0<-ua.b2, dst1<-ua.b3, dst2<-ub.b2, dst3<-ub.b3
  return __builtin_amdgcn_perm(ub, ua, 0x07060302u);
}

__device__ __forceinline__ __bf16 f2bf(float f) {
  unsigned u = __builtin_bit_cast(unsigned, f);
  u += 0x7FFFu + ((u >> 16) & 1u);           // RNE (cold path only)
  unsigned short h = (unsigned short)(u >> 16);
  return __builtin_bit_cast(__bf16, h);
}

// A fragment (16x32 bf16, MxK): lane m=lane&15, half=lane>>4,
// element i -> K = (i&7) + 8*half + 16*(i>=8). base points at row m, K = kb+8*half.
__device__ __forceinline__ v16bf load_a_lds(const __bf16* base) {
  v8bf lo = *(const v8bf*)(base);
  v8bf hi = *(const v8bf*)(base + 16);
  v16bf a;
#pragma unroll
  for (int i = 0; i < 8; ++i) { a[i] = lo[i]; a[i + 8] = hi[i]; }
  return a;
}

// B fragment (32x16 bf16, KxN): lane n=lane&15 selects weight row; element i ->
// K=(i&7)+8*half+16*(i>=8). p = row at K=8*half; loads p[0..7], p[16..23], packed to bf16.
__device__ __forceinline__ v16bf load_wrow_bf16(const float* __restrict__ p) {
  const float4 q0 = *(const float4*)(p + 0);
  const float4 q1 = *(const float4*)(p + 4);
  const float4 q2 = *(const float4*)(p + 16);
  const float4 q3 = *(const float4*)(p + 20);
  v8i w;
  w[0] = (int)pk_bf16(q0.x, q0.y); w[1] = (int)pk_bf16(q0.z, q0.w);
  w[2] = (int)pk_bf16(q1.x, q1.y); w[3] = (int)pk_bf16(q1.z, q1.w);
  w[4] = (int)pk_bf16(q2.x, q2.y); w[5] = (int)pk_bf16(q2.z, q2.w);
  w[6] = (int)pk_bf16(q3.x, q3.y); w[7] = (int)pk_bf16(q3.z, q3.w);
  return __builtin_bit_cast(v16bf, w);
}

__device__ __forceinline__ v8f wmma_bf16(v16bf a, v16bf b, v8f c) {
  return __builtin_amdgcn_wmma_f32_16x16x32_bf16(false, a, false, b,
                                                 (short)0, c, false, false);
}

__global__ __launch_bounds__(256)
void moe_swiglu_fused(const float* __restrict__ x,
                      const float* __restrict__ wg,
                      const float* __restrict__ wu,
                      const float* __restrict__ wd,
                      float* __restrict__ out) {
  __shared__ __bf16 Xs[kMT * kXS];       // 66048 B : X tile (32 x 1024) bf16
  __shared__ __bf16 Hs[2][kMT * kHS];    // 2x 8704 B : hidden chunk, double-buffered

  const int e   = (int)blockIdx.x >> 1;
  const int m0  = ((int)blockIdx.x & 1) * kMT;
  const int tid = (int)threadIdx.x;
  const int wv  = tid >> 5;
  const int ln  = tid & 31;
  const int n   = ln & 15;   // fragment column / weight-row select
  const int hf  = ln >> 4;   // lane half

  // ---- Phase 0: stage X tile as bf16 into LDS -----------------------------
  const float* xg = x + (size_t)e * kC * kD + (size_t)m0 * kD;
  for (int i = tid; i < kMT * kD / 4; i += 256) {
    const int el = i * 4;
    const int r = el >> 10;          // /kD
    const int c = el & (kD - 1);
    const float4 v = *(const float4*)(xg + el);
    v2i b4;
    b4[0] = (int)pk_bf16(v.x, v.y);
    b4[1] = (int)pk_bf16(v.z, v.w);
    *(v2i*)&Xs[r * kXS + c] = b4;
  }
  __syncthreads();

  // Persistent output accumulators: 2 M-tiles x 8 N-tiles of D per wave.
  v8f acc[2][8];
#pragma unroll
  for (int t = 0; t < 2; ++t)
#pragma unroll
    for (int d = 0; d < 8; ++d)
#pragma unroll
      for (int j = 0; j < 8; ++j) acc[t][d][j] = 0.0f;

  const float* wge = wg + (size_t)e * kDF * kD;
  const float* wue = wu + (size_t)e * kDF * kD;
  const float* wde = wd + (size_t)e * kD * kDF;

  for (int ch = 0; ch < kDF / kFC; ++ch) {
    __bf16* hsb = &Hs[ch & 1][0];

    // ---- Phase A: gate & up GEMMs for this wave's 16 f-columns ------------
    const int f0 = ch * kFC + wv * 16;
    const float* wgrow = wge + (size_t)(f0 + n) * kD + 8 * hf;
    const float* wurow = wue + (size_t)(f0 + n) * kD + 8 * hf;

    v8f hg0, hg1, hu0, hu1;
#pragma unroll
    for (int j = 0; j < 8; ++j) { hg0[j] = 0.f; hg1[j] = 0.f; hu0[j] = 0.f; hu1[j] = 0.f; }

#pragma unroll 4
    for (int ks = 0; ks < kD / 32; ++ks) {
      const int kb = ks * 32;
      const v16bf a0 = load_a_lds(&Xs[n * kXS + kb + 8 * hf]);
      const v16bf a1 = load_a_lds(&Xs[(16 + n) * kXS + kb + 8 * hf]);
      const v16bf bg = load_wrow_bf16(wgrow + kb);
      const v16bf bu = load_wrow_bf16(wurow + kb);
      hg0 = wmma_bf16(a0, bg, hg0);
      hg1 = wmma_bf16(a1, bg, hg1);
      hu0 = wmma_bf16(a0, bu, hu0);
      hu1 = wmma_bf16(a1, bu, hu1);
    }

    // silu(gate)*up -> bf16 hidden chunk in LDS. D-frag element j -> row j+8*hf.
#pragma unroll
    for (int j = 0; j < 8; ++j) {
      {
        const float g = hg0[j], u = hu0[j];
        const float h = g * __builtin_amdgcn_rcpf(1.0f + __expf(-g)) * u;
        hsb[(j + 8 * hf) * kHS + wv * 16 + n] = f2bf(h);
      }
      {
        const float g = hg1[j], u = hu1[j];
        const float h = g * __builtin_amdgcn_rcpf(1.0f + __expf(-g)) * u;
        hsb[(16 + j + 8 * hf) * kHS + wv * 16 + n] = f2bf(h);
      }
    }
    __syncthreads();   // writers done; (double-buffering covers WAR two chunks out)

    // ---- Phase B: out += hidden_chunk @ Wd[:, chunk]^T --------------------
#pragma unroll
    for (int ks = 0; ks < kFC / 32; ++ks) {
      const int kb = ks * 32;
      const v16bf a0 = load_a_lds(&hsb[n * kHS + kb + 8 * hf]);
      const v16bf a1 = load_a_lds(&hsb[(16 + n) * kHS + kb + 8 * hf]);
#pragma unroll
      for (int dt = 0; dt < 8; ++dt) {
        const int dcol = (wv * 8 + dt) * 16;
        const float* wdrow =
            wde + (size_t)(dcol + n) * kDF + (size_t)ch * kFC + kb + 8 * hf;
        const v16bf bd = load_wrow_bf16(wdrow);
        acc[0][dt] = wmma_bf16(a0, bd, acc[0][dt]);
        acc[1][dt] = wmma_bf16(a1, bd, acc[1][dt]);
      }
    }
  }

  // ---- Epilogue: spill f32 accumulators to global -------------------------
  float* oe = out + (size_t)e * kC * kD + (size_t)m0 * kD;
#pragma unroll
  for (int t = 0; t < 2; ++t)
#pragma unroll
    for (int dt = 0; dt < 8; ++dt) {
      const int dcol = (wv * 8 + dt) * 16 + n;
#pragma unroll
      for (int j = 0; j < 8; ++j)
        oe[(size_t)(t * 16 + j + 8 * hf) * kD + dcol] = acc[t][dt][j];
    }
}

extern "C" void kernel_launch(void* const* d_in, const int* in_sizes, int n_in,
                              void* d_out, int out_size, void* d_ws, size_t ws_size,
                              hipStream_t stream) {
  (void)in_sizes; (void)n_in; (void)d_ws; (void)ws_size; (void)out_size;
  const float* x  = (const float*)d_in[0];
  const float* wg = (const float*)d_in[1];
  const float* wu = (const float*)d_in[2];
  const float* wd = (const float*)d_in[3];
  float* o = (float*)d_out;
  dim3 grid(kE * 2), block(256);
  hipLaunchKernelGGL(moe_swiglu_fused, grid, block, 0, stream, x, wg, wu, wd, o);
}
